// RateCodedSNNBaseline_83468394430526
// MI455X (gfx1250) — matrix operand order repaired and checked by reading, not compiled, not run on hardware
//
#include <hip/hip_runtime.h>

// Rate-coded SNN, fused for MI455X (gfx1250, wave32, WMMA bf16).
// B=16384, D=2048, H=256, C=1000 (padded 1024), T=25.

#define BATCH     16384
#define D_INPUT   2048
#define HIDDEN    256
#define NCLS      1000
#define NCLS_PAD  1024
#define NSTEPS    25

typedef __attribute__((ext_vector_type(16))) __bf16   v16bf;
typedef __attribute__((ext_vector_type(8)))  float    v8f;
typedef __attribute__((ext_vector_type(4)))  unsigned v4u;

union BFrag { v16bf bf; v4u u[2]; };

__device__ __forceinline__ unsigned short f2bf(float f) {
  union { float f; unsigned u; } v; v.f = f;
  unsigned r = v.u + 0x7FFFu + ((v.u >> 16) & 1u);   // round-to-nearest-even
  return (unsigned short)(r >> 16);
}

__device__ __forceinline__ float fsigmoid(float z) {
  // 1/(1+exp(-z)) = rcp(1 + exp2(-z*log2e)) -> v_exp_f32 + v_rcp_f32
  float e = __builtin_amdgcn_exp2f(z * -1.442695041f);
  return __builtin_amdgcn_rcpf(1.0f + e);
}

// ---- W1 pack: B[k][n]=W1[n][k] -> bf16 B-frag linear (64 kfrags x 16 ntiles)
__global__ __launch_bounds__(256) void snn_pack_w1(const float* __restrict__ W1,
                                                   unsigned short* __restrict__ w1p) {
  unsigned e   = blockIdx.x * 256u + threadIdx.x;        // < 524288
  unsigned kk  = e & 15u;
  unsigned ln  = (e >> 4) & 31u;
  unsigned ng  = (e >> 9) & 15u;
  unsigned kf  = e >> 13;                                 // 0..63
  unsigned n   = ng * 16u + (ln & 15u);
  unsigned k   = kf * 32u + (ln >> 4) * 16u + kk;
  w1p[e] = f2bf(W1[n * D_INPUT + k]);
}

// ---- W2 pack: padded to 1024 classes (8 kfrags x 64 ntiles); also zeroes the
// sparsity accumulator slot in d_out (harness poisons d_out).
__global__ __launch_bounds__(256) void snn_pack_w2(const float* __restrict__ W2,
                                                   unsigned short* __restrict__ w2p,
                                                   float* __restrict__ out) {
  unsigned e   = blockIdx.x * 256u + threadIdx.x;        // < 262144
  unsigned kk  = e & 15u;
  unsigned ln  = (e >> 4) & 31u;
  unsigned ng  = (e >> 9) & 63u;
  unsigned kf  = e >> 15;                                 // 0..7
  unsigned c   = ng * 16u + (ln & 15u);
  unsigned k   = kf * 32u + (ln >> 4) * 16u + kk;
  w2p[e] = (c < NCLS) ? f2bf(W2[c * HIDDEN + k]) : (unsigned short)0;
  if (e == 0) out[(size_t)BATCH * NCLS] = 0.0f;
}

// ---- Fused SNN kernel: 512 threads = 16 waves; WG owns 32 batch rows.
__global__ __launch_bounds__(512) void snn_fused(const float* __restrict__ x,
                                                 const float* __restrict__ b1,
                                                 const float* __restrict__ b2,
                                                 const unsigned short* __restrict__ w1p,
                                                 const unsigned short* __restrict__ w2p,
                                                 float* __restrict__ out) {
  __shared__ unsigned short smem[8192];   // 16 KB A-frag staging (x chunk / spk1)
  __shared__ float red;

  const int tid     = threadIdx.x;
  const int lane    = tid & 31;
  const int w       = tid >> 5;     // wave 0..15
  const int mt      = w >> 3;       // M-tile (rows mt*16..mt*16+15)
  const int o       = w & 7;        // hidden/class octant
  const int l16     = lane & 15;
  const int lhalf   = lane >> 4;
  const int rowbase = blockIdx.x * 32;

  if (tid == 0) red = 0.0f;

  const float bv1a = b1[o * 32 + l16];
  const float bv1b = b1[o * 32 + 16 + l16];
  float b2v[8];
#pragma unroll
  for (int nt = 0; nt < 8; ++nt) {
    int c = o * 128 + nt * 16 + l16;
    b2v[nt] = (c < NCLS) ? b2[c] : 0.0f;
  }

  // ---------- Phase 1: cur1 = x @ W1^T + b1 (computed once) ----------
  v8f cur1[2];
#pragma unroll
  for (int i = 0; i < 8; ++i) { cur1[0][i] = bv1a; cur1[1][i] = bv1b; }

  // cooperative x loader mapping: 16 threads per row, 8 f32 per thread per pass
  const int lr  = tid >> 4;          // row 0..31
  const int lg  = tid & 15;          // k-group
  const int lmt = lr >> 4;
  const int lrr = lr & 15;
  const float* xrow = x + (size_t)(rowbase + lr) * D_INPUT;

  for (int ch = 0; ch < 8; ++ch) {   // K chunks of 256
#pragma unroll
    for (int pass = 0; pass < 2; ++pass) {
      int k = lg * 8 + pass * 128;                       // 8-aligned, < 256
      const float4* s4 = (const float4*)(xrow + ch * 256 + k);
      float4 fa = s4[0], fb = s4[1];
      v4u val;
      val.x = (unsigned)f2bf(fa.x) | ((unsigned)f2bf(fa.y) << 16);
      val.y = (unsigned)f2bf(fa.z) | ((unsigned)f2bf(fa.w) << 16);
      val.z = (unsigned)f2bf(fb.x) | ((unsigned)f2bf(fb.y) << 16);
      val.w = (unsigned)f2bf(fb.z) | ((unsigned)f2bf(fb.w) << 16);
      int kfl    = k >> 5;
      int kl     = k & 31;
      int lane_a = lrr + 16 * (kl >> 4);                 // A-matrix lane swizzle
      int idx    = ((kfl * 2 + lmt) * 32 + lane_a) * 16 + ((kl >> 3) & 1) * 8;
      *(v4u*)&smem[idx] = val;
    }
    __syncthreads();
#pragma unroll
    for (int kfl = 0; kfl < 8; ++kfl) {
      BFrag a;
      const v4u* ap = (const v4u*)&smem[((kfl * 2 + mt) * 32 + lane) * 16];
      a.u[0] = ap[0]; a.u[1] = ap[1];
      int kf = ch * 8 + kfl;
#pragma unroll
      for (int t = 0; t < 2; ++t) {
        BFrag bb;
        const v4u* bp = (const v4u*)&w1p[(size_t)((kf * 16 + (o * 2 + t)) * 32 + lane) * 16];
        bb.u[0] = bp[0]; bb.u[1] = bp[1];
        cur1[t] = __builtin_amdgcn_wmma_f32_16x16x32_bf16(
            false, a.bf, false, bb.bf, (short)0, cur1[t], false, false);
      }
    }
    __syncthreads();
  }

  // ---------- Phase 2: 25 LIF steps, GEMM2 on WMMA bf16 ----------
  v8f m2[8], spike[8], mem1v[2];
#pragma unroll
  for (int nt = 0; nt < 8; ++nt)
#pragma unroll
    for (int i = 0; i < 8; ++i) { m2[nt][i] = 0.0f; spike[nt][i] = 0.0f; }
#pragma unroll
  for (int t = 0; t < 2; ++t)
#pragma unroll
    for (int i = 0; i < 8; ++i) mem1v[t][i] = 0.0f;

  float ssum = 0.0f;

  for (int step = 0; step < NSTEPS; ++step) {
    // layer-1 LIF + surrogate spike; scatter spk1(bf16) into LDS A-frag layout
#pragma unroll
    for (int t = 0; t < 2; ++t) {
#pragma unroll
      for (int i = 0; i < 8; ++i) {
        float m = 0.9f * mem1v[t][i] + cur1[t][i];
        float s = fsigmoid(10.0f * (m - 1.0f));
        mem1v[t][i] = m * (1.0f - s);
        if (step == NSTEPS - 1) ssum += s;
        int lane_a = (i + 8 * lhalf) + 16 * t;
        int idx    = ((o * 2 + mt) * 32 + lane_a) * 16 + l16;
        smem[idx] = f2bf(s);
      }
    }
    __syncthreads();

    // mem2 pre-accumulate: 0.9*mem2 + b2, then WMMA-accumulate spk1 @ W2^T
#pragma unroll
    for (int nt = 0; nt < 8; ++nt)
#pragma unroll
      for (int i = 0; i < 8; ++i)
        m2[nt][i] = 0.9f * m2[nt][i] + b2v[nt];

#pragma unroll
    for (int kf = 0; kf < 8; ++kf) {
      BFrag a;
      const v4u* ap = (const v4u*)&smem[((kf * 2 + mt) * 32 + lane) * 16];
      a.u[0] = ap[0]; a.u[1] = ap[1];
#pragma unroll
      for (int nt = 0; nt < 8; ++nt) {
        BFrag bb;
        const v4u* bp = (const v4u*)&w2p[(size_t)((kf * 64 + o * 8 + nt) * 32 + lane) * 16];
        bb.u[0] = bp[0]; bb.u[1] = bp[1];
        m2[nt] = __builtin_amdgcn_wmma_f32_16x16x32_bf16(
            false, a.bf, false, bb.bf, (short)0, m2[nt], false, false);
      }
    }
    __syncthreads();   // spk1 LDS consumed; next step may overwrite

    // layer-2 LIF + spike accumulation
#pragma unroll
    for (int nt = 0; nt < 8; ++nt) {
#pragma unroll
      for (int i = 0; i < 8; ++i) {
        float m = m2[nt][i];
        float s = fsigmoid(10.0f * (m - 1.0f));
        spike[nt][i] += s;
        m2[nt][i] = m * (1.0f - s);
      }
    }
  }

  // ---------- Epilogue: write spike_count/25 and sparsity ----------
  const float inv_steps = 1.0f / (float)NSTEPS;
#pragma unroll
  for (int nt = 0; nt < 8; ++nt) {
    int c = o * 128 + nt * 16 + l16;
    if (c < NCLS) {
#pragma unroll
      for (int i = 0; i < 8; ++i) {
        int b = rowbase + mt * 16 + i + 8 * lhalf;
        out[(size_t)b * NCLS + c] = spike[nt][i] * inv_steps;
      }
    }
  }

  atomicAdd(&red, ssum);
  __syncthreads();
  if (tid == 0)
    atomicAdd(&out[(size_t)BATCH * NCLS],
              red * (1.0f / ((float)BATCH * (float)HIDDEN)));
}

extern "C" void kernel_launch(void* const* d_in, const int* in_sizes, int n_in,
                              void* d_out, int out_size, void* d_ws, size_t ws_size,
                              hipStream_t stream) {
  (void)in_sizes; (void)n_in; (void)out_size; (void)ws_size;
  const float* x  = (const float*)d_in[0];
  const float* W1 = (const float*)d_in[1];
  const float* b1 = (const float*)d_in[2];
  const float* W2 = (const float*)d_in[3];
  const float* b2 = (const float*)d_in[4];
  float* out = (float*)d_out;

  unsigned short* w2p = (unsigned short*)d_ws;           // 262144 bf16 (512 KB)
  unsigned short* w1p = w2p + 262144;                    // 524288 bf16 (1 MB)

  snn_pack_w2<<<1024, 256, 0, stream>>>(W2, w2p, out);
  snn_pack_w1<<<2048, 256, 0, stream>>>(W1, w1p);
  snn_fused<<<BATCH / 32, 512, 0, stream>>>(x, b1, b2, w1p, w2p, out);
}
